// BasicAttentionModule_66408784331549
// MI455X (gfx1250) — matrix operand order repaired and checked by reading, not compiled, not run on hardware
//
#include <hip/hip_runtime.h>

typedef __attribute__((ext_vector_type(16))) _Float16 v16h;
typedef __attribute__((ext_vector_type(8)))  float    v8f;

#define B_  4
#define C_  256
#define H_  256
#define W_  256
#define S_  8
#define HP_ 32
#define WP_ 32
#define N_  1024
#define KC_ 32

// ---------------------------------------------------------------------------
// 1) AvgPool 8x8 -> f16 tokens X[B, C, N]   (N = 32*32, token n = hp*32+wp)
// ---------------------------------------------------------------------------
__global__ void pool_kernel(const float* __restrict__ f, _Float16* __restrict__ X) {
  int idx = blockIdx.x * blockDim.x + threadIdx.x;        // [0, B*C*N)
  if (idx >= B_ * C_ * N_) return;
  int wp = idx & (WP_ - 1);
  int hp = (idx >> 5) & (HP_ - 1);
  int bc = idx >> 10;                                     // b*C + c
  const float* base = f + ((size_t)bc * H_ + hp * S_) * W_ + wp * S_;
  float s = 0.f;
#pragma unroll
  for (int r = 0; r < S_; ++r) {
    const float4* p4 = (const float4*)(base + r * W_);    // 32B-aligned
    float4 a = p4[0], b = p4[1];
    s += a.x + a.y + a.z + a.w + b.x + b.y + b.z + b.w;
  }
  X[idx] = (_Float16)(s * (1.f / 64.f));
}

// ---------------------------------------------------------------------------
// 2) Y[b][m,n] = sum_c W[m,c] * X[b][c,n] + bias[m]   (f16 out, f32 acc WMMA)
//    grid = (N/16, M/16, B), 1 wave/block; K-loop over C in 32-chunks.
// ---------------------------------------------------------------------------
__global__ void gemm_wxb_kernel(const float* __restrict__ Wt, const float* __restrict__ bias,
                                const _Float16* __restrict__ X, _Float16* __restrict__ Y, int M) {
  int lane = threadIdx.x;
  int row = lane & 15;          // A row / B col / D col within tile
  int hi  = lane >> 4;
  int n0 = blockIdx.x * 16, m0 = blockIdx.y * 16, b = blockIdx.z;
  const _Float16* Xb = X + (size_t)b * C_ * N_;
  v8f acc = {};
  for (int k0 = 0; k0 < C_; k0 += 32) {
    v16h a, bb;
#pragma unroll
    for (int i = 0; i < 16; ++i) {
      int ka = (i & 7) + ((i >> 3) << 4) + hi * 8;        // ISA A-frag K index
      a[i] = (_Float16)Wt[(size_t)(m0 + row) * C_ + k0 + ka];
      int kb = (i & 7) + ((i >> 3) << 4) + hi * 8;        // ISA B-frag K index
      bb[i] = Xb[(size_t)(k0 + kb) * N_ + n0 + row];
    }
    acc = __builtin_amdgcn_wmma_f32_16x16x32_f16(false, a, false, bb,
                                                 (short)0, acc, false, false);
  }
  _Float16* Yb = Y + (size_t)b * M * N_;
#pragma unroll
  for (int r = 0; r < 8; ++r) {
    int m = m0 + r + hi * 8;                              // ISA C/D row mapping
    Yb[(size_t)m * N_ + n0 + row] = (_Float16)(acc[r] + bias[m]);
  }
}

// ---------------------------------------------------------------------------
// 3+4 fused) For one 16-row block n0..n0+15 of one batch:
//    energy row-block (16 x 1024) via WMMA (one 16x16x32 op per m-tile)
//    staged in LDS (64 KB of the WGP's 320 KB), then row softmax with
//    wave32 shfl-xor reductions, writing f16 probs P[b][n,m] directly.
//    Removes the 16 MB fp32 energy round-trip through HBM/L2.
//    grid = (N/16, B), block = 256 threads (8 waves).
// ---------------------------------------------------------------------------
__global__ void energy_softmax_kernel(const _Float16* __restrict__ Q,
                                      const _Float16* __restrict__ Km,
                                      _Float16* __restrict__ P) {
  __shared__ float Es[16][N_];                            // 64 KB
  int n0 = blockIdx.x * 16, b = blockIdx.y;
  int t = threadIdx.x;
  int wave = t >> 5, lane = t & 31;
  int row = lane & 15, hi = lane >> 4;
  const _Float16* Qb = Q  + (size_t)b * KC_ * N_;
  const _Float16* Kb = Km + (size_t)b * KC_ * N_;
  const float scale = 0.17677669529663687f;               // 32^-0.5

  // A fragment: A[row=n, k] = Q[k, n0+row]  (shared across this wave's tiles)
  v16h a;
#pragma unroll
  for (int i = 0; i < 16; ++i) {
    int ka = (i & 7) + ((i >> 3) << 4) + hi * 8;
    a[i] = Qb[(size_t)ka * N_ + n0 + row];
  }
  // Each wave covers 8 of the 64 m-tiles.
#pragma unroll
  for (int j = 0; j < 8; ++j) {
    int m0 = (wave * 8 + j) * 16;
    v16h bb;
#pragma unroll
    for (int i = 0; i < 16; ++i)
      bb[i] = Kb[(size_t)((i & 7) + ((i >> 3) << 4) + hi * 8) * N_ + m0 + row];
    v8f acc = {};
    acc = __builtin_amdgcn_wmma_f32_16x16x32_f16(false, a, false, bb,
                                                 (short)0, acc, false, false);
#pragma unroll
    for (int r = 0; r < 8; ++r)
      Es[r + hi * 8][m0 + row] = acc[r] * scale;
  }
  __syncthreads();

  // Row softmax: row = t>>4, 16 threads per row, 64 columns each (stride 16).
  int r = t >> 4, sub = t & 15;
  float m = -3.4e38f;
#pragma unroll 8
  for (int k = 0; k < 64; ++k) m = fmaxf(m, Es[r][sub + 16 * k]);
#pragma unroll
  for (int msk = 8; msk >= 1; msk >>= 1) m = fmaxf(m, __shfl_xor(m, msk, 32));
  float s = 0.f;
#pragma unroll 8
  for (int k = 0; k < 64; ++k) {
    float e = __expf(Es[r][sub + 16 * k] - m);
    Es[r][sub + 16 * k] = e;
    s += e;
  }
#pragma unroll
  for (int msk = 8; msk >= 1; msk >>= 1) s += __shfl_xor(s, msk, 32);
  float inv = 1.0f / s;
  _Float16* p = P + ((size_t)b * N_ + n0 + r) * N_;
#pragma unroll 8
  for (int k = 0; k < 64; ++k)
    p[sub + 16 * k] = (_Float16)(Es[r][sub + 16 * k] * inv);
}

// ---------------------------------------------------------------------------
// 5) O[b][c,n] = sum_m V[b][c,m] * P[b][n,m]   (K = 1024 -> 32 WMMAs/tile)
//    grid = (N/16, C/16, B)
// ---------------------------------------------------------------------------
__global__ void outgemm_kernel(const _Float16* __restrict__ V, const _Float16* __restrict__ P,
                               float* __restrict__ O) {
  int lane = threadIdx.x;
  int row = lane & 15, hi = lane >> 4;
  int n0 = blockIdx.x * 16, c0 = blockIdx.y * 16, b = blockIdx.z;
  const _Float16* Vb = V + (size_t)b * C_ * N_;
  const _Float16* Pb = P + (size_t)b * N_ * N_;
  v8f acc = {};
  for (int k0 = 0; k0 < N_; k0 += 32) {
    v16h a, bb;
#pragma unroll
    for (int i = 0; i < 16; ++i) {
      int ka = (i & 7) + ((i >> 3) << 4) + hi * 8;
      a[i]  = Vb[(size_t)(c0 + row) * N_ + k0 + ka];      // A[c, m] (contig. chunks)
      int kb = (i & 7) + ((i >> 3) << 4) + hi * 8;
      bb[i] = Pb[(size_t)(n0 + row) * N_ + k0 + kb];      // B[m, n] = P[n, m] (contig.)
    }
    acc = __builtin_amdgcn_wmma_f32_16x16x32_f16(false, a, false, bb,
                                                 (short)0, acc, false, false);
  }
  float* Ob = O + (size_t)b * C_ * N_;
#pragma unroll
  for (int r = 0; r < 8; ++r)
    Ob[(size_t)(c0 + r + hi * 8) * N_ + n0 + row] = acc[r];
}

// ---------------------------------------------------------------------------
// 6) out = repeat8x8(O) + features   (bandwidth-dominant; float4 vectorized)
//    One thread per (b, c, h, wp) -> 8 consecutive outputs.
// ---------------------------------------------------------------------------
__global__ void upsample_add_kernel(const float* __restrict__ O, const float* __restrict__ f,
                                    float* __restrict__ out) {
  int idx = blockIdx.x * blockDim.x + threadIdx.x;        // [0, B*C*H*WP)
  if (idx >= B_ * C_ * H_ * WP_) return;
  int wp = idx & (WP_ - 1);
  int h  = (idx >> 5) & (H_ - 1);
  int bc = idx >> 13;
  float o = O[(size_t)bc * N_ + (h >> 3) * WP_ + wp];
  size_t base = ((size_t)bc * H_ + h) * W_ + wp * 8;
  const float4* fp = (const float4*)(f + base);
  float4* op = (float4*)(out + base);
  float4 a = fp[0], b = fp[1];
  a.x += o; a.y += o; a.z += o; a.w += o;
  b.x += o; b.y += o; b.z += o; b.w += o;
  op[0] = a; op[1] = b;
}

// ---------------------------------------------------------------------------
extern "C" void kernel_launch(void* const* d_in, const int* in_sizes, int n_in,
                              void* d_out, int out_size, void* d_ws, size_t ws_size,
                              hipStream_t stream) {
  const float* feat = (const float*)d_in[0];
  const float* q_w  = (const float*)d_in[1];
  const float* q_b  = (const float*)d_in[2];
  const float* k_w  = (const float*)d_in[3];
  const float* k_b  = (const float*)d_in[4];
  const float* v_w  = (const float*)d_in[5];
  const float* v_b  = (const float*)d_in[6];
  float* out = (float*)d_out;

  // Workspace carve-up (~16.5 MB total; fp32 energy buffer eliminated)
  _Float16* X  = (_Float16*)d_ws;                       // [B,C,N]  f16  2 MB
  _Float16* Q  = X  + (size_t)B_ * C_  * N_;            // [B,K,N]  f16  256 KB
  _Float16* Km = Q  + (size_t)B_ * KC_ * N_;            // [B,K,N]  f16  256 KB
  _Float16* V  = Km + (size_t)B_ * KC_ * N_;            // [B,C,N]  f16  2 MB
  _Float16* P  = V  + (size_t)B_ * C_  * N_;            // [B,N,N]  f16  8 MB
  float*    O  = (float*)(P + (size_t)B_ * N_ * N_);    // [B,C,N]  f32  4 MB

  pool_kernel<<<(B_ * C_ * N_ + 255) / 256, 256, 0, stream>>>(feat, X);

  gemm_wxb_kernel<<<dim3(N_ / 16, KC_ / 16, B_), 32, 0, stream>>>(q_w, q_b, X, Q,  KC_);
  gemm_wxb_kernel<<<dim3(N_ / 16, KC_ / 16, B_), 32, 0, stream>>>(k_w, k_b, X, Km, KC_);
  gemm_wxb_kernel<<<dim3(N_ / 16, C_  / 16, B_), 32, 0, stream>>>(v_w, v_b, X, V,  C_);

  energy_softmax_kernel<<<dim3(N_ / 16, B_), 256, 0, stream>>>(Q, Km, P);
  outgemm_kernel<<<dim3(N_ / 16, C_ / 16, B_), 32, 0, stream>>>(V, P, O);

  upsample_add_kernel<<<(B_ * C_ * H_ * WP_ + 255) / 256, 256, 0, stream>>>(O, feat, out);
}
